// GraphVampNet_63342177681977
// MI455X (gfx1250) — compile-verified, matrix operand order checked by reading, not run on hardware
//
#include <hip/hip_runtime.h>

typedef __bf16 v16bf __attribute__((ext_vector_type(16)));
typedef float  v8f   __attribute__((ext_vector_type(8)));

static constexpr int  BB = 32, NN = 512, MM = 32;
static constexpr int  HA = 64, HB = 101, CH = 128, KP = 256;
static constexpr int  NCONV = 4;
static constexpr long ROWS = (long)BB * NN * MM;   // 524288
static constexpr int  NBN  = BB * NN;              // 16384
static constexpr float BN_EPS = 1e-5f;

__device__ __forceinline__ unsigned short f2bf(float f) {
  unsigned u = __float_as_uint(f);
  u += 0x7FFFu + ((u >> 16) & 1u);
  return (unsigned short)(u >> 16);
}
__device__ __forceinline__ float bf2f(unsigned short h) {
  return __uint_as_float(((unsigned)h) << 16);
}

union FragBF { v16bf v; uint4 q[2]; };

// ---------------------------------------------------------------------------
// Prep kernels
// ---------------------------------------------------------------------------
__global__ void k_nbr(const float* __restrict__ dist, unsigned short* __restrict__ nbr_emb) {
  long i = (long)blockIdx.x * 256 + threadIdx.x;   // ROWS*128
  long row = i >> 7; int k = (int)(i & 127);
  unsigned short v = 0;
  if (k < HB) {
    float d = dist[row] - 0.1f * (float)k;
    v = f2bf(__expf(-d * d * 100.0f));             // 1/STEP^2 = 100
  }
  nbr_emb[i] = v;
}

__global__ void k_atom0(const float* __restrict__ emb, float* __restrict__ atom) {
  long i = (long)blockIdx.x * 256 + threadIdx.x;   // NBN*HA
  atom[i] = emb[i % ((long)NN * HA)];
}

__global__ void k_wt(const float* __restrict__ fcw, unsigned short* __restrict__ wt) {
  long i = (long)blockIdx.x * 256 + threadIdx.x;   // NCONV*128*256, layout [l][col][k]
  int l = (int)(i >> 15);
  int rem = (int)(i & 32767);
  int c = rem >> 8, k = rem & 255;
  wt[i] = (k < 2 * HA + HB) ? f2bf(fcw[((long)l * (2 * HA + HB) + k) * CH + c])
                            : (unsigned short)0;
}

// ---------------------------------------------------------------------------
// Main WMMA GEMM: [128 rows x 256 K] x [256 K x 128 ch] per block
// dynamic LDS = 128*256*2 = 65536 bytes (A tile, reused for stats)
// ---------------------------------------------------------------------------
__global__ void __launch_bounds__(256)
k_gemm(const float* __restrict__ atom,
       const int* __restrict__ nbr_idx,
       const unsigned short* __restrict__ nbr_emb,   // [ROWS][128] bf16
       const unsigned short* __restrict__ wt,        // [128 col][256 K] bf16 (layer slice)
       unsigned short* __restrict__ y,               // [ROWS][128] bf16 out
       float* __restrict__ part1)                    // [grid][256] sums / sumsq
{
  extern __shared__ char smem[];
  unsigned short* sA = (unsigned short*)smem;        // [128][256]
  const int t = threadIdx.x;
  const int lane = t & 31, wave = t >> 5;
  const long rowBase = (long)blockIdx.x * 128;

  // ---- build A tile: [self(64) | nbr(64) | basis(101 pad 128)] ----
  {
    const int r = t >> 1;            // 0..127
    const int half = t & 1;
    const long g = rowBase + r;
    const int b = (int)(g / ((long)NN * MM));
    const int n = (int)((g / MM) % NN);
    if (half == 0) {
      const float* selfrow = atom + ((long)b * NN + n) * HA;
      const int nb = nbr_idx[g];
      const float* nbrrow = atom + ((long)b * NN + nb) * HA;
      unsigned* dst32 = (unsigned*)(sA + r * KP);
      for (int k = 0; k < HA; k += 2) {
        dst32[k >> 1] = (unsigned)f2bf(selfrow[k]) | ((unsigned)f2bf(selfrow[k + 1]) << 16);
        dst32[(HA + k) >> 1] = (unsigned)f2bf(nbrrow[k]) | ((unsigned)f2bf(nbrrow[k + 1]) << 16);
      }
    } else {
      const uint4* src = (const uint4*)(nbr_emb + g * 128);
      uint4* dst = (uint4*)(sA + r * KP + 128);
      for (int j = 0; j < 16; ++j) dst[j] = src[j];
    }
  }
  __syncthreads();

  // ---- WMMA main loop: each wave -> 16 rows x 128 cols ----
  v8f acc[8];
  {
    v8f z = {0.f, 0.f, 0.f, 0.f, 0.f, 0.f, 0.f, 0.f};
    for (int i = 0; i < 8; ++i) acc[i] = z;
  }
  const int rA = wave * 16 + (lane & 15);
  const int khalf = ((lane >> 4) & 1) << 3;          // 0 or 8
  for (int kc = 0; kc < 8; ++kc) {
    const int k0 = kc * 32 + khalf;
    FragBF a;
    a.q[0] = *(const uint4*)(sA + rA * KP + k0);
    a.q[1] = *(const uint4*)(sA + rA * KP + k0 + 16);
    for (int nt = 0; nt < 8; ++nt) {
      FragBF bf;
      const unsigned short* wcol = wt + (long)(nt * 16 + (lane & 15)) * KP + k0;
      bf.q[0] = *(const uint4*)(wcol);
      bf.q[1] = *(const uint4*)(wcol + 16);
      acc[nt] = __builtin_amdgcn_wmma_f32_16x16x32_bf16(
          false, a.v, false, bf.v, (short)0, acc[nt], false, false);
    }
  }

  // ---- store y (bf16) + accumulate per-lane channel stats ----
  const int rowOff = ((lane >> 4) & 1) << 3;         // +8 rows for upper lanes
  float lsum[8], lsq[8];
  for (int nt = 0; nt < 8; ++nt) { lsum[nt] = 0.f; lsq[nt] = 0.f; }
  for (int nt = 0; nt < 8; ++nt) {
    const int col = nt * 16 + (lane & 15);
    for (int v = 0; v < 8; ++v) {
      float x = acc[nt][v];
      long rr = rowBase + wave * 16 + rowOff + v;
      y[rr * CH + col] = f2bf(x);
      lsum[nt] += x; lsq[nt] += x * x;
    }
  }
  __syncthreads();                                   // sA no longer needed
  float* sSum = (float*)smem;
  float* sSq = sSum + CH;
  if (t < CH) { sSum[t] = 0.f; sSq[t] = 0.f; }
  __syncthreads();
  for (int nt = 0; nt < 8; ++nt) {
    const int col = nt * 16 + (lane & 15);
    atomicAdd(&sSum[col], lsum[nt]);
    atomicAdd(&sSq[col], lsq[nt]);
  }
  __syncthreads();
  if (t < CH) {
    part1[(long)blockIdx.x * 256 + t]      = sSum[t];
    part1[(long)blockIdx.x * 256 + CH + t] = sSq[t];
  }
}

// ---------------------------------------------------------------------------
// BN1 stats finalize: mean/var over ROWS per channel
// ---------------------------------------------------------------------------
__global__ void k_stats1(const float* __restrict__ part1, float* __restrict__ stats1, int nblk) {
  int h = threadIdx.x;                               // 0..127
  float s = 0.f, s2 = 0.f;
  for (int b = 0; b < nblk; ++b) {
    s  += part1[(long)b * 256 + h];
    s2 += part1[(long)b * 256 + CH + h];
  }
  float inv = 1.0f / (float)ROWS;
  float mu = s * inv;
  float var = s2 * inv - mu * mu;
  stats1[h] = mu;
  stats1[CH + h] = rsqrtf(var + BN_EPS);
}

// ---------------------------------------------------------------------------
// BN1 apply + sigmoid(filt)*relu(core), sum over M -> summed [NBN][64]
// ---------------------------------------------------------------------------
__global__ void __launch_bounds__(256)
k_gate(const unsigned short* __restrict__ y, const float* __restrict__ stats1,
       const float* __restrict__ g1, const float* __restrict__ b1,
       float* __restrict__ summed)
{
  int t = threadIdx.x;
  int h = t & 63;
  long bn = (long)blockIdx.x * 4 + (t >> 6);
  long rowbase = bn * MM;
  float muF = stats1[h],      rsF = stats1[CH + h];
  float muC = stats1[64 + h], rsC = stats1[CH + 64 + h];
  float gF = g1[h], bF = b1[h], gC = g1[64 + h], bC = b1[64 + h];
  float acc = 0.f;
  for (int m = 0; m < MM; ++m) {
    long r = rowbase + m;
    float xf = (bf2f(y[r * CH + h])      - muF) * rsF * gF + bF;
    float xc = (bf2f(y[r * CH + 64 + h]) - muC) * rsC * gC + bC;
    float sg = 1.0f / (1.0f + __expf(-xf));
    float rl = xc > 0.f ? xc : 0.f;
    acc += sg * rl;
  }
  summed[bn * HA + h] = acc;
}

// ---------------------------------------------------------------------------
// BN2 stats: one block per channel (64 blocks)
// ---------------------------------------------------------------------------
__global__ void __launch_bounds__(256)
k_stats2(const float* __restrict__ summed, float* __restrict__ stats2) {
  __shared__ float sS[256], sQ[256];
  int h = blockIdx.x;
  float s = 0.f, s2 = 0.f;
  for (int r = threadIdx.x; r < NBN; r += 256) {
    float x = summed[(long)r * HA + h];
    s += x; s2 += x * x;
  }
  sS[threadIdx.x] = s; sQ[threadIdx.x] = s2;
  __syncthreads();
  for (int o = 128; o > 0; o >>= 1) {
    if (threadIdx.x < o) { sS[threadIdx.x] += sS[threadIdx.x + o]; sQ[threadIdx.x] += sQ[threadIdx.x + o]; }
    __syncthreads();
  }
  if (threadIdx.x == 0) {
    float inv = 1.0f / (float)NBN;
    float mu = sS[0] * inv;
    float var = sQ[0] * inv - mu * mu;
    stats2[h] = mu; stats2[64 + h] = rsqrtf(var + BN_EPS);
  }
}

// ---------------------------------------------------------------------------
// BN2 apply + residual relu
// ---------------------------------------------------------------------------
__global__ void k_resid(float* __restrict__ atom, const float* __restrict__ summed,
                        const float* __restrict__ stats2,
                        const float* __restrict__ g2, const float* __restrict__ b2) {
  long i = (long)blockIdx.x * 256 + threadIdx.x;     // NBN*HA
  int h = (int)(i & 63);
  float x = atom[i] + (summed[i] - stats2[h]) * stats2[64 + h] * g2[h] + b2[h];
  atom[i] = x > 0.f ? x : 0.f;
}

// ---------------------------------------------------------------------------
// Head: relu, mean pool over N, classifier, softmax
// ---------------------------------------------------------------------------
__global__ void k_head(const float* __restrict__ atom, const float* __restrict__ cw,
                       const float* __restrict__ cb, float* __restrict__ out) {
  __shared__ float pooled[64];
  __shared__ float logit[10];
  int b = blockIdx.x, h = threadIdx.x;               // 64 threads
  float s = 0.f;
  const float* base = atom + (long)b * NN * HA;
  for (int n = 0; n < NN; ++n) {
    float x = base[(long)n * HA + h];
    s += x > 0.f ? x : 0.f;
  }
  pooled[h] = s / (float)NN;
  __syncthreads();
  if (h < 10) {
    float l = cb[h];
    for (int k = 0; k < 64; ++k) l += pooled[k] * cw[k * 10 + h];
    logit[h] = l;
  }
  __syncthreads();
  if (h == 0) {
    float mx = logit[0];
    for (int c = 1; c < 10; ++c) mx = fmaxf(mx, logit[c]);
    float e[10]; float ssum = 0.f;
    for (int c = 0; c < 10; ++c) { e[c] = __expf(logit[c] - mx); ssum += e[c]; }
    for (int c = 0; c < 10; ++c) out[b * 10 + c] = e[c] / ssum;
  }
}

// ---------------------------------------------------------------------------
extern "C" void kernel_launch(void* const* d_in, const int* in_sizes, int n_in,
                              void* d_out, int out_size, void* d_ws, size_t ws_size,
                              hipStream_t stream) {
  const float* dist = (const float*)d_in[0];
  const int*   idx  = (const int*)d_in[1];
  const float* emb  = (const float*)d_in[2];
  const float* fcw  = (const float*)d_in[3];
  // d_in[4] = fc_b: unused — a per-channel constant is cancelled by training-mode BN
  const float* bn1g = (const float*)d_in[5];
  const float* bn1b = (const float*)d_in[6];
  const float* bn2g = (const float*)d_in[7];
  const float* bn2b = (const float*)d_in[8];
  const float* cw   = (const float*)d_in[9];
  const float* cb   = (const float*)d_in[10];
  float* out = (float*)d_out;

  char* ws = (char*)d_ws;
  size_t off = 0;
  unsigned short* nbr_emb = (unsigned short*)(ws + off); off += (size_t)ROWS * 128 * 2;
  unsigned short* wt      = (unsigned short*)(ws + off); off += (size_t)NCONV * 128 * 256 * 2;
  float* atom   = (float*)(ws + off); off += (size_t)NBN * HA * 4;
  float* summed = (float*)(ws + off); off += (size_t)NBN * HA * 4;
  unsigned short* y = (unsigned short*)(ws + off); off += (size_t)ROWS * CH * 2;
  float* part1  = (float*)(ws + off); off += (size_t)4096 * 256 * 4;
  float* stats1 = (float*)(ws + off); off += 256 * 4;
  float* stats2 = (float*)(ws + off); off += 128 * 4;
  (void)off; (void)ws_size; (void)in_sizes; (void)n_in; (void)out_size;

  k_nbr  <<< (int)(ROWS * 128 / 256), 256, 0, stream >>>(dist, nbr_emb);
  k_atom0<<< (int)(NBN * HA / 256),   256, 0, stream >>>(emb, atom);
  k_wt   <<< (NCONV * 128 * 256) / 256, 256, 0, stream >>>(fcw, wt);

  const int GEMM_BLOCKS = (int)(ROWS / 128);         // 4096
  for (int l = 0; l < NCONV; ++l) {
    k_gemm<<<GEMM_BLOCKS, 256, 128 * KP * 2, stream>>>(
        atom, idx, nbr_emb, wt + (size_t)l * 128 * 256, y, part1);
    k_stats1<<<1, 128, 0, stream>>>(part1, stats1, GEMM_BLOCKS);
    k_gate<<<NBN / 4, 256, 0, stream>>>(y, stats1, bn1g + l * CH, bn1b + l * CH, summed);
    k_stats2<<<HA, 256, 0, stream>>>(summed, stats2);
    k_resid<<< (int)(NBN * HA / 256), 256, 0, stream>>>(
        atom, summed, stats2, bn2g + l * HA, bn2b + l * HA);
  }
  k_head<<<BB, 64, 0, stream>>>(atom, cw, cb, out);
}
